// CustomNeighborhoodAttention3D_65678639891301
// MI455X (gfx1250) — matrix operand order repaired and checked by reading, not compiled, not run on hardware
//
#include <hip/hip_runtime.h>
#include <math.h>

typedef __attribute__((ext_vector_type(16))) _Float16 v16h;
typedef __attribute__((ext_vector_type(8)))  float    v8f;

#define DD 5
#define HH 40
#define WW 80
#define CC 128
#define NH 4
#define HD 32
#define NVOX  (DD*HH*WW)     // 16000
#define NVOXP (NVOX + 64)    // padded so W-overreach (<= +31) stays in-bounds
#define SCALE 0.17677669529663687f   // 32^-0.5

// A-operand (16x32 f16): lane holds row ln; K = e + 16*hi
__device__ __forceinline__ int a_k(int e, int hi){ return e + (hi << 4); }
// B-operand (32x16 f16): lane holds column ln; K = e + 16*hi
__device__ __forceinline__ int b_k(int e, int hi){ return e + (hi << 4); }

__global__ void cvt_f32_f16(const float* __restrict__ src, _Float16* __restrict__ dst, int n){
  int i = blockIdx.x * blockDim.x + threadIdx.x;
  if (i < n) dst[i] = (_Float16)src[i];
}

// dst[n][K] = (f16) src[k][n]  -- column-major weights so B-fragments are per-lane contiguous
__global__ void cvt_transpose_f16(const float* __restrict__ src, _Float16* __restrict__ dst, int K, int N){
  int i = blockIdx.x * blockDim.x + threadIdx.x;
  if (i < K * N){
    int k = i / N, n = i % N;
    dst[n * K + k] = (_Float16)src[i];
  }
}

__global__ void zero_f16(_Float16* __restrict__ p, int n){
  int i = blockIdx.x * blockDim.x + threadIdx.x;
  if (i < n) p[i] = (_Float16)0.f;
}

// qkv = x @ w_qkv + b_qkv ; q scaled; q,k -> [vox][CC] f16, v -> transposed vT[CC][NVOXP]
__global__ void __launch_bounds__(256)
qkv_gemm(const _Float16* __restrict__ xh, const _Float16* __restrict__ wT /*[384][128]*/,
         const float* __restrict__ bias,
         _Float16* __restrict__ qh, _Float16* __restrict__ kh, _Float16* __restrict__ vT){
  int wid  = (blockIdx.x * blockDim.x + threadIdx.x) >> 5;
  const int nRT = NVOX / 16;                 // 1000 row tiles
  if (wid >= nRT * 24) return;
  int rt = wid / 24, ct = wid % 24;
  int lane = threadIdx.x & 31;
  int hi = lane >> 4, ln = lane & 15;
  int row0 = rt * 16, col0 = ct * 16;

  v8f acc = {};
  #pragma unroll
  for (int kc = 0; kc < 4; ++kc){
    v16h A, B;
    #pragma unroll
    for (int e = 0; e < 16; ++e){
      A[e] = xh[(row0 + ln) * CC + kc * 32 + a_k(e, hi)];
      B[e] = wT[(col0 + ln) * CC + kc * 32 + b_k(e, hi)];   // contiguous per lane
    }
    acc = __builtin_amdgcn_wmma_f32_16x16x32_f16(false, A, false, B, (short)0, acc, false, false);
  }

  int col = col0 + ln;
  float bv = bias[col];
  int sec = ct >> 3;                         // wave-uniform: 0=q, 1=k, 2=v
  int c   = col & (CC - 1);
  float mul = (sec == 0) ? SCALE : 1.0f;
  if (sec == 2){
    #pragma unroll
    for (int r = 0; r < 8; ++r){             // consecutive voxels -> vectorizable store
      int m = r + hi * 8;
      vT[(size_t)c * NVOXP + row0 + m] = (_Float16)(acc[r] + bv);
    }
  } else {
    _Float16* dst = (sec == 0) ? qh : kh;
    #pragma unroll
    for (int r = 0; r < 8; ++r){
      int m = r + hi * 8;
      dst[(row0 + m) * CC + c] = (_Float16)((acc[r] + bv) * mul);
    }
  }
}

// flash-style neighborhood attention: one wave per (d, h, 16-wide W strip, head)
// candidate keys = 9 (dz,hy) rows x 32 consecutive W voxels (20 real + 12 masked pad)
__global__ void __launch_bounds__(256)
na3d_attn(const _Float16* __restrict__ qh, const _Float16* __restrict__ kh,
          const _Float16* __restrict__ vT, _Float16* __restrict__ oh){
  __shared__ _Float16 smem[8 * 16 * 32];     // 8 KB: per-wave P chunk staging
  int wid = (blockIdx.x * blockDim.x + threadIdx.x) >> 5;
  if (wid >= NH * 5 * HH * DD) return;       // 4000 waves exactly
  int lane = threadIdx.x & 31;
  int hi = lane >> 4, ln = lane & 15;
  _Float16* lds = smem + (threadIdx.x >> 5) * (16 * 32);

  int head = wid & 3;
  int t1 = wid >> 2;
  int wt = t1 % 5;  t1 /= 5;
  int h  = t1 % HH;
  int d  = t1 / HH;

  int w0 = wt * 16;
  int wbase  = min(max(w0 - 2, 0), WW - 20);
  int dstart = min(max(d  - 1, 0), DD - 3);
  int hstart = min(max(h  - 1, 0), HH - 3);
  int qvox0 = (d * HH + h) * WW + w0;

  // Q fragment (A layout): row = ln
  v16h QA;
  #pragma unroll
  for (int e = 0; e < 16; ++e)
    QA[e] = qh[(qvox0 + ln) * CC + head * HD + a_k(e, hi)];

  v8f O0 = {}, O1 = {};
  float runmax[8], runsum[8];
  #pragma unroll
  for (int r = 0; r < 8; ++r){ runmax[r] = -3.0e38f; runsum[r] = 0.f; }

  for (int c = 0; c < 9; ++c){
    int dz = c / 3, hy = c % 3;
    int rowvox = ((dstart + dz) * HH + (hstart + hy)) * WW + wbase;

    if (c < 8){                               // prefetch next key row
      int dz2 = (c + 1) / 3, hy2 = (c + 1) % 3;
      int nxt = ((dstart + dz2) * HH + (hstart + hy2)) * WW + wbase;
      __builtin_prefetch(&kh[(size_t)nxt * CC + head * HD], 0, 0);
      __builtin_prefetch(&vT[(size_t)(head * HD + ln) * NVOXP + nxt], 0, 0);
    }

    // logits for 32 keys of this row: 2 WMMA tiles
    v8f S0, S1;
    {
      v16h KB;
      int kvox = rowvox + ln;                 // column key: wz = ln
      #pragma unroll
      for (int e = 0; e < 16; ++e) KB[e] = kh[(size_t)kvox * CC + head * HD + b_k(e, hi)];
      v8f z = {};
      S0 = __builtin_amdgcn_wmma_f32_16x16x32_f16(false, QA, false, KB, (short)0, z, false, false);
    }
    {
      v16h KB;
      int kvox = rowvox + 16 + ln;            // column key: wz = 16 + ln
      #pragma unroll
      for (int e = 0; e < 16; ++e) KB[e] = kh[(size_t)kvox * CC + head * HD + b_k(e, hi)];
      v8f z = {};
      S1 = __builtin_amdgcn_wmma_f32_16x16x32_f16(false, QA, false, KB, (short)0, z, false, false);
    }

    // masked online-softmax update; row m = r + 8*hi lives in one 16-lane half-group
    #pragma unroll
    for (int r = 0; r < 8; ++r){
      int m = r + hi * 8;
      int w = w0 + m;
      int wsrt = min(max(w - 2, 0), WW - 5);
      int kwa0 = wbase + ln;
      int kwa1 = wbase + 16 + ln;
      bool v0 = (kwa0 >= wsrt) && (kwa0 <= wsrt + 4);
      bool v1 = (ln < 4) && (kwa1 >= wsrt) && (kwa1 <= wsrt + 4);
      float s0 = v0 ? S0[r] : -3.0e38f;
      float s1 = v1 ? S1[r] : -3.0e38f;
      float cm = fmaxf(s0, s1);
      #pragma unroll
      for (int off = 1; off < 16; off <<= 1)
        cm = fmaxf(cm, __shfl_xor(cm, off, 32));
      float nm = fmaxf(runmax[r], cm);
      float alpha = __expf(runmax[r] - nm);
      float p0 = v0 ? __expf(s0 - nm) : 0.f;
      float p1 = v1 ? __expf(s1 - nm) : 0.f;
      float ps = p0 + p1;
      #pragma unroll
      for (int off = 1; off < 16; off <<= 1)
        ps += __shfl_xor(ps, off, 32);
      runsum[r] = runsum[r] * alpha + ps;
      runmax[r] = nm;
      O0[r] *= alpha;
      O1[r] *= alpha;
      lds[m * 32 + ln]      = (_Float16)p0;   // stage P chunk (C layout -> LDS)
      lds[m * 32 + 16 + ln] = (_Float16)p1;
    }
    asm volatile("s_wait_dscnt 0" ::: "memory");  // wave-local LDS RAW fence

    // P A-fragment: one contiguous 16-half run -> ds b128 loads
    v16h PA;
    #pragma unroll
    for (int e = 0; e < 16; ++e)
      PA[e] = lds[ln * 32 + a_k(e, hi)];

    // V B-fragments from vT: 16 consecutive voxels per lane -> b128 loads
    {
      v16h VB;
      #pragma unroll
      for (int e = 0; e < 16; ++e)
        VB[e] = vT[(size_t)(head * HD + ln) * NVOXP + rowvox + (hi << 4) + e];
      O0 = __builtin_amdgcn_wmma_f32_16x16x32_f16(false, PA, false, VB, (short)0, O0, false, false);
    }
    {
      v16h VB;
      #pragma unroll
      for (int e = 0; e < 16; ++e)
        VB[e] = vT[(size_t)(head * HD + 16 + ln) * NVOXP + rowvox + (hi << 4) + e];
      O1 = __builtin_amdgcn_wmma_f32_16x16x32_f16(false, PA, false, VB, (short)0, O1, false, false);
    }
  }

  #pragma unroll
  for (int r = 0; r < 8; ++r){
    int m = r + hi * 8;
    float inv = 1.f / runsum[r];
    oh[(qvox0 + m) * CC + head * HD +      ln] = (_Float16)(O0[r] * inv);
    oh[(qvox0 + m) * CC + head * HD + 16 + ln] = (_Float16)(O1[r] * inv);
  }
}

// out = attn_out @ w_proj + b_proj  (f32 output)
__global__ void __launch_bounds__(256)
proj_gemm(const _Float16* __restrict__ ohb, const _Float16* __restrict__ wT /*[128][128]*/,
          const float* __restrict__ bias, float* __restrict__ out){
  int wid = (blockIdx.x * blockDim.x + threadIdx.x) >> 5;
  const int nRT = NVOX / 16;
  if (wid >= nRT * 8) return;
  int rt = wid / 8, ct = wid % 8;
  int lane = threadIdx.x & 31;
  int hi = lane >> 4, ln = lane & 15;
  int row0 = rt * 16, col0 = ct * 16;

  v8f acc = {};
  #pragma unroll
  for (int kc = 0; kc < 4; ++kc){
    v16h A, B;
    #pragma unroll
    for (int e = 0; e < 16; ++e){
      A[e] = ohb[(row0 + ln) * CC + kc * 32 + a_k(e, hi)];
      B[e] = wT[(col0 + ln) * CC + kc * 32 + b_k(e, hi)];
    }
    acc = __builtin_amdgcn_wmma_f32_16x16x32_f16(false, A, false, B, (short)0, acc, false, false);
  }
  int col = col0 + ln;
  float bv = bias[col];
  #pragma unroll
  for (int r = 0; r < 8; ++r){
    int m = r + hi * 8;
    out[(row0 + m) * CC + col] = acc[r] + bv;
  }
}

extern "C" void kernel_launch(void* const* d_in, const int* in_sizes, int n_in,
                              void* d_out, int out_size, void* d_ws, size_t ws_size,
                              hipStream_t stream){
  (void)in_sizes; (void)n_in; (void)out_size; (void)ws_size;
  const float* x      = (const float*)d_in[0];
  const float* w_qkv  = (const float*)d_in[1];
  const float* b_qkv  = (const float*)d_in[2];
  const float* w_proj = (const float*)d_in[3];
  const float* b_proj = (const float*)d_in[4];
  float* out = (float*)d_out;

  char* ws = (char*)d_ws;
  size_t off = 0;
  auto alloc = [&](size_t bytes)->void* {
    void* p = ws + off;
    off += (bytes + 255) & ~(size_t)255;
    return p;
  };
  _Float16* xh     = (_Float16*)alloc((size_t)NVOX * CC * 2);
  _Float16* wqkvT  = (_Float16*)alloc((size_t)(3 * CC) * CC * 2);   // [384][128]
  _Float16* wprojT = (_Float16*)alloc((size_t)CC * CC * 2);         // [128][128]
  _Float16* qh     = (_Float16*)alloc((size_t)NVOX  * CC * 2);
  _Float16* kh     = (_Float16*)alloc((size_t)NVOXP * CC * 2);      // padded rows
  _Float16* vT     = (_Float16*)alloc((size_t)CC * NVOXP * 2);      // transposed, padded
  _Float16* ohb    = (_Float16*)alloc((size_t)NVOX * CC * 2);

  int nx = NVOX * CC;
  cvt_f32_f16<<<(nx + 255) / 256, 256, 0, stream>>>(x, xh, nx);
  cvt_transpose_f16<<<(CC * 3 * CC + 255) / 256, 256, 0, stream>>>(w_qkv, wqkvT, CC, 3 * CC);
  cvt_transpose_f16<<<(CC * CC + 255) / 256, 256, 0, stream>>>(w_proj, wprojT, CC, CC);
  // zero padded regions so masked reads stay deterministic/finite
  zero_f16<<<(64 * CC + 255) / 256, 256, 0, stream>>>(kh + (size_t)NVOX * CC, 64 * CC);
  zero_f16<<<(CC * NVOXP + 255) / 256, 256, 0, stream>>>(vT, CC * NVOXP);

  // 1000 row tiles * 24 col tiles = 24000 waves -> 3000 blocks of 8 waves
  qkv_gemm<<<3000, 256, 0, stream>>>(xh, wqkvT, b_qkv, qh, kh, vT);
  // 5*40*5*4 = 4000 waves -> 500 blocks
  na3d_attn<<<500, 256, 0, stream>>>(qh, kh, vT, ohb);
  // 1000 * 8 = 8000 waves -> 1000 blocks
  proj_gemm<<<1000, 256, 0, stream>>>(ohb, wprojT, b_proj, out);
}
